// MSDT_47545287967267
// MI455X (gfx1250) — matrix operand order repaired and checked by reading, not compiled, not run on hardware
//
#include <hip/hip_runtime.h>

// ---------------------------------------------------------------------------
// MI455X (gfx1250). Bandwidth-bound conv pipeline:
//  - All convs = implicit GEMM on v_wmma_f32_16x16x32_bf16 (f32 accumulate).
//  - Internal activations bf16 NHWC (halves HBM traffic vs fp32; channels-
//    innermost => contiguous LDS staging + single 16B store epilogue).
//  - Non-upsampled convs: DOUBLE-BUFFERED async global->LDS staging
//    (global_load_async_to_lds_b128 / ASYNCcnt): chunk k+1 DMAs into one LDS
//    buffer while the 9 WMMAs of chunk k run from the other. One
//    s_wait_asynccnt + one barrier per chunk.
//  - Upsampled convs fuse bilinear 2x interp into a VALU staging loop.
//  - Per-wave D tile 16 couts x 16 pixels; workgroup 8 waves = 32 couts x
//    (2 rows x 32 cols). K loop = (Cin/32) stages x ksz^2 WMMAs per stage.
// ---------------------------------------------------------------------------

typedef __attribute__((ext_vector_type(16))) __bf16 v16bf;
typedef __attribute__((ext_vector_type(8)))  __bf16 v8bf;
typedef __attribute__((ext_vector_type(8)))  float  v8f;
typedef int v4i __attribute__((vector_size(16)));   // matches builtin prototype

#if __has_builtin(__builtin_amdgcn_global_load_async_to_lds_b128) && \
    __has_builtin(__builtin_amdgcn_s_wait_asynccnt)
#define HAVE_ASYNC_LDS 1
#else
#define HAVE_ASYNC_LDS 0
#endif

// AS1 (global) / AS3 (LDS) qualified vector pointers for the async builtin
#define ASG(p) ((__attribute__((address_space(1))) v4i*)(p))
#define ASL(p) ((__attribute__((address_space(3))) v4i*)(p))

// fp32 NCHW plane bilinear (tail kernels)
__device__ __forceinline__ float bilin_sample(const float* __restrict__ plane,
                                              int oy, int ox, int inH, int inW,
                                              int outH, int outW) {
  float py = oy * ((float)(inH - 1) / (float)(outH - 1));
  float px = ox * ((float)(inW - 1) / (float)(outW - 1));
  int y0 = (int)py, x0 = (int)px;
  int y1 = y0 + 1; if (y1 > inH - 1) y1 = inH - 1;
  int x1 = x0 + 1; if (x1 > inW - 1) x1 = inW - 1;
  float ty = py - (float)y0, tx = px - (float)x0;
  float a = plane[(size_t)y0 * inW + x0], b = plane[(size_t)y0 * inW + x1];
  float c = plane[(size_t)y1 * inW + x0], d = plane[(size_t)y1 * inW + x1];
  float top = a + (b - a) * tx, bot = c + (d - c) * tx;
  return top + (bot - top) * ty;
}

// bf16 NHWC bilinear: base points at (b,0,0,ch); pixel stride = C
__device__ __forceinline__ float bilin_nhwc(const __bf16* __restrict__ base,
                                            int oy, int ox, int inH, int inW, int C,
                                            int outH, int outW) {
  float py = oy * ((float)(inH - 1) / (float)(outH - 1));
  float px = ox * ((float)(inW - 1) / (float)(outW - 1));
  int y0 = (int)py, x0 = (int)px;
  int y1 = y0 + 1; if (y1 > inH - 1) y1 = inH - 1;
  int x1 = x0 + 1; if (x1 > inW - 1) x1 = inW - 1;
  float ty = py - (float)y0, tx = px - (float)x0;
  float a = (float)base[((size_t)y0 * inW + x0) * C];
  float b = (float)base[((size_t)y0 * inW + x1) * C];
  float c = (float)base[((size_t)y1 * inW + x0) * C];
  float d = (float)base[((size_t)y1 * inW + x1) * C];
  float top = a + (b - a) * tx, bot = c + (d - c) * tx;
  return top + (bot - top) * ty;
}

// Fused conv: outH(bf16 NHWC at [ctot,coff]) or outF(fp32 NCHW) =
//   BN/bias( conv_ksz( in_or_up2x(in bf16 NHWC) ) ) (+ residual bf16 NHWC)
__global__ __launch_bounds__(256) void conv_wmma(
    const __bf16* __restrict__ in, const __bf16* __restrict__ wp,
    const float* __restrict__ gam, const float* __restrict__ bet,
    const float* __restrict__ mea, const float* __restrict__ vr,
    const __bf16* __restrict__ resid,
    __bf16* __restrict__ outH, float* __restrict__ outF,
    int Bn, int Cin, int Cout, int H, int W, int inH, int inW,
    int up, int ksz, int pad, int ctot, int coff, int hasBN)
{
  __shared__ __attribute__((aligned(64))) __bf16 patch[2][4 * 34 * 32];

  const int tid  = threadIdx.x, lane = tid & 31, wave = tid >> 5;
  const int mblk = wave & 1, nblk = wave >> 1;
  const int tileX = blockIdx.x * 32, tileY = blockIdx.y * 2;
  const int cpb = Cout >> 5;
  const int bb  = blockIdx.z / cpb;
  const int cb  = (blockIdx.z % cpb) << 5;
  const int prows = 1 + ksz, pcols = 31 + ksz;
  const int taps  = ksz * ksz;

  v8f acc;
  #pragma unroll
  for (int i = 0; i < 8; ++i) acc[i] = 0.f;

  const int m   = lane & 15;
  const int kbA = (lane < 16) ? 0 : 16;
  const int kbB = (lane < 16) ? 0 : 16;
  const int nChunks = Cin >> 5;
  const __bf16* inb = in + (size_t)bb * inH * inW * Cin;   // batch base (NHWC)

  // copy-staging for one 32-channel chunk into LDS buffer `buf`
  auto stage_copy = [&](int kc, int buf) {
    const int cinB = kc << 5;
    const int tot4 = prows * pcols * 4;          // unit = 16B = 8 channels
    for (int u = tid; u < tot4; u += 256) {
      int sub = u & 3; int rc = u >> 2;
      int col = rc % pcols, row = rc / pcols;
      int iy = tileY + row - pad, ix = tileX + col - pad;
      __bf16* lp = &patch[buf][((row * pcols + col) << 5) + sub * 8];
      if (iy >= 0 && iy < H && ix >= 0 && ix < W) {
        const __bf16* gp = inb + ((size_t)iy * inW + ix) * Cin + cinB + sub * 8;
#if HAVE_ASYNC_LDS
        __builtin_amdgcn_global_load_async_to_lds_b128(ASG(gp), ASL(lp), 0, 0);
#else
        *(v8bf*)lp = *(const v8bf*)gp;
#endif
      } else {
        v8bf z;
        #pragma unroll
        for (int i = 0; i < 8; ++i) z[i] = (__bf16)0.f;
        *(v8bf*)lp = z;
      }
    }
  };

  if (!up) stage_copy(0, 0);                     // prologue: fill buffer 0

  for (int kc = 0; kc < nChunks; ++kc) {
    const int cinB = kc << 5;
    int buf;
    if (up) {
      // single-buffer VALU staging with fused bilinear upsample
      buf = 0;
      __syncthreads();                           // previous stage consumed
      const int tot = prows * pcols * 32;
      for (int e = tid; e < tot; e += 256) {
        int ch = e & 31; int rc = e >> 5;
        int col = rc % pcols, row = rc / pcols;
        int iy = tileY + row - pad, ix = tileX + col - pad;
        float val = 0.f;
        if (iy >= 0 && iy < H && ix >= 0 && ix < W)
          val = bilin_nhwc(inb + cinB + ch, iy, ix, inH, inW, Cin, H, W);
        patch[0][e] = (__bf16)val;
      }
      __syncthreads();
      if (kc + 1 < nChunks)
        __builtin_prefetch(inb + cinB + 32, 0, 1);
    } else {
      // double-buffered async pipeline:
      //  wait for current buffer, then kick off DMA of the next one
      buf = kc & 1;
#if HAVE_ASYNC_LDS
      __builtin_amdgcn_s_wait_asynccnt(0);
#endif
      __syncthreads();   // (a) buf data visible, (b) buf^1 free to overwrite
      if (kc + 1 < nChunks) stage_copy(kc + 1, buf ^ 1);
    }

    for (int t = 0; t < taps; ++t) {
      int r = t / ksz, s = t - r * ksz;
      // A fragment: bf16 weights [tap][cout][cin]
      const __bf16* wrow = wp + ((size_t)t * Cout + (cb + mblk * 16 + m)) * (size_t)Cin + cinB;
      v8bf alo = *(const v8bf*)(wrow + kbA);
      v8bf ahi = *(const v8bf*)(wrow + 8 + kbA);
      v16bf afrag = __builtin_shufflevector(alo, ahi,
          0, 1, 2, 3, 4, 5, 6, 7, 8, 9, 10, 11, 12, 13, 14, 15);
      // B fragment: LDS [row][col][ch]; lane = pixel, 16 contiguous channels
      int prow = (nblk >> 1) + r;
      int pcol = (nblk & 1) * 16 + m + s;
      const v16bf* bp = (const v16bf*)&patch[buf][((prow * pcols + pcol) << 5) + kbB];
      v16bf bfrag = *bp;
      acc = __builtin_amdgcn_wmma_f32_16x16x32_bf16(false, afrag, false, bfrag,
                                                    (short)0, acc, false, false);
    }
  }

  // Epilogue: lane holds col n, 8 contiguous channels (base..base+7)
  const int orow = tileY + (nblk >> 1);
  const int ocol = tileX + (nblk & 1) * 16 + m;
  const int base = cb + mblk * 16 + ((lane < 16) ? 0 : 8);
  float y[8];
  #pragma unroll
  for (int vv = 0; vv < 8; ++vv) {
    int c = base + vv;
    float sc, sh;
    if (hasBN) { sc = gam[c] * rsqrtf(vr[c] + 1e-5f); sh = bet[c] - mea[c] * sc; }
    else       { sc = 1.f;                            sh = bet[c]; }
    y[vv] = acc[vv] * sc + sh;
  }
  if (resid) {
    const v8bf rv = *(const v8bf*)&resid[(((size_t)bb * H + orow) * W + ocol) * Cout + base];
    #pragma unroll
    for (int vv = 0; vv < 8; ++vv) y[vv] += (float)rv[vv];
  }
  if (outF) {   // fp32 NCHW (final feature output)
    #pragma unroll
    for (int vv = 0; vv < 8; ++vv)
      outF[(((size_t)bb * Cout + base + vv) * H + orow) * W + ocol] = y[vv];
  } else {      // bf16 NHWC, one 16B store
    v8bf pk;
    #pragma unroll
    for (int vv = 0; vv < 8; ++vv) pk[vv] = (__bf16)y[vv];
    *(v8bf*)&outH[(((size_t)bb * H + orow) * W + ocol) * ctot + coff + base] = pk;
  }
}

// OIHW fp32 -> [tap][cout][cin] bf16
__global__ void repack_w(const float* __restrict__ w, __bf16* __restrict__ wp,
                         int Cout, int Cin, int ksq, long tot) {
  long i = (long)blockIdx.x * blockDim.x + threadIdx.x;
  if (i >= tot) return;
  int ci = (int)(i % Cin);
  long r = i / Cin;
  int co = (int)(r % Cout);
  int t  = (int)(r / Cout);
  wp[i] = (__bf16)w[((size_t)co * Cin + ci) * ksq + t];
}

// fp32 NCHW -> bf16 NHWC
__global__ void k_tobf16_nhwc(const float* __restrict__ src, __bf16* __restrict__ dst,
                              int C, int HW, long tot) {
  long i = (long)blockIdx.x * blockDim.x + threadIdx.x;
  if (i >= tot) return;
  int c = (int)(i % C);
  long r = i / C;
  int p = (int)(r % HW);
  int b = (int)(r / HW);
  dst[i] = (__bf16)src[((size_t)b * C + c) * HW + p];
}

__global__ void ew_mul_add_h(const __bf16* __restrict__ a, const __bf16* __restrict__ b,
                             const __bf16* __restrict__ r, __bf16* __restrict__ o, long n) {
  long i = (long)blockIdx.x * blockDim.x + threadIdx.x;
  if (i < n) o[i] = (__bf16)((float)a[i] * (float)b[i] + (float)r[i]);
}

// NHWC channel-concat copy
__global__ void copy_ch_h(const __bf16* __restrict__ src, __bf16* __restrict__ dst,
                          int Cs, int Ctot, int coff, int HW, long tot) {
  long i = (long)blockIdx.x * blockDim.x + threadIdx.x;
  if (i >= tot) return;
  int c = (int)(i % Cs);
  long r = i / Cs;
  int p = (int)(r % HW);
  int b = (int)(r / HW);
  dst[((size_t)b * HW + p) * Ctot + coff + c] = src[i];
}

// 1x1 conv C->1 + bias (reads fp32 NCHW fused_features)
__global__ void k_conv6(const float* __restrict__ in, const float* __restrict__ w,
                        const float* __restrict__ bias, float* __restrict__ out,
                        int HW, int Cin, long n) {
  long i = (long)blockIdx.x * blockDim.x + threadIdx.x;
  if (i >= n) return;
  int p = (int)(i % HW);
  int bb = (int)(i / HW);
  const float* base = in + (size_t)bb * Cin * HW + p;
  float acc = bias[0];
  for (int c = 0; c < Cin; ++c) acc += base[(size_t)c * HW] * w[c];
  out[i] = acc;
}

__global__ void k_upconvin(const float* __restrict__ ff, const float* __restrict__ w,
                           const float* __restrict__ bias, float* __restrict__ X2, long n) {
  long i = (long)blockIdx.x * blockDim.x + threadIdx.x;
  if (i >= n) return;
  int p = (int)(i % 65536);
  int bb = (int)(i / 65536);
  int y = p >> 8, x = p & 255;
  const float* plane = ff + (size_t)bb * 16384;
  float acc = bias[0];
  #pragma unroll
  for (int r = 0; r < 3; ++r)
    #pragma unroll
    for (int s = 0; s < 3; ++s) {
      int iy = y + r - 1, ix = x + s - 1;
      if (iy < 0 || iy >= 256 || ix < 0 || ix >= 256) continue;
      acc += w[r * 3 + s] * bilin_sample(plane, iy, ix, 128, 128, 256, 256);
    }
  X2[i] = acc;
}

__global__ void k_haar(const float* __restrict__ X2, const float* __restrict__ ff1,
                       float* __restrict__ fused, long n) {
  long i = (long)blockIdx.x * blockDim.x + threadIdx.x;
  if (i >= n) return;
  int p = (int)(i % 16384);
  int bb = (int)(i / 16384);
  int yo = p >> 7, xo = p & 127;
  const float* Xp = X2 + (size_t)bb * 65536;
  float a  = Xp[(2 * yo) * 256 + 2 * xo],     b2 = Xp[(2 * yo) * 256 + 2 * xo + 1];
  float c  = Xp[(2 * yo + 1) * 256 + 2 * xo], d  = Xp[(2 * yo + 1) * 256 + 2 * xo + 1];
  float ll = (a + b2 + c + d) * 0.5f, lh = (a + b2 - c - d) * 0.5f;
  float hl = (a - b2 + c - d) * 0.5f, hh = (a - b2 - c + d) * 0.5f;
  float edge = fabsf(lh) + fabsf(hl) + fabsf(hh);
  fused[((size_t)bb * 2) * 16384 + p]     = ll;
  fused[((size_t)bb * 2 + 1) * 16384 + p] = ff1[i] - edge;
}

__global__ void k_fusion(const float* __restrict__ fused, const float* __restrict__ fw,
                         const float* __restrict__ fb, float* __restrict__ out, long n) {
  long i = (long)blockIdx.x * blockDim.x + threadIdx.x;
  if (i >= n) return;
  int p = (int)(i % 16384);
  int bb = (int)(i / 16384);
  int y = p >> 7, x = p & 127;
  float acc = fb[0];
  for (int ch = 0; ch < 2; ++ch)
    for (int r = 0; r < 3; ++r)
      for (int s = 0; s < 3; ++s) {
        int iy = y + r - 1, ix = x + s - 1;
        if (iy < 0 || iy >= 128 || ix < 0 || ix >= 128) continue;
        acc += fw[(ch * 3 + r) * 3 + s] * fused[((size_t)bb * 2 + ch) * 16384 + iy * 128 + ix];
      }
  out[i] = acc;
}

extern "C" void kernel_launch(void* const* d_in, const int* in_sizes, int n_in,
                              void* d_out, int out_size, void* d_ws, size_t ws_size,
                              hipStream_t stream) {
  (void)in_sizes; (void)n_in; (void)out_size; (void)ws_size;
  const int Bn = 16, C0 = 64;
  const float* x1 = (const float*)d_in[0];
  const float* x2 = (const float*)d_in[1];
  const float* x3 = (const float*)d_in[2];
  const float* x4 = (const float*)d_in[3];
  struct BC { const float *w, *g, *b, *m, *v; };
  auto P = [&](int i) {
    return BC{(const float*)d_in[i], (const float*)d_in[i + 1], (const float*)d_in[i + 2],
              (const float*)d_in[i + 3], (const float*)d_in[i + 4]};
  };
  BC pconv1 = P(4), pconv2 = P(9), pconv3 = P(14), pconv7 = P(19), pconv4 = P(24),
     pconv5 = P(29), pconv6 = P(34), pcc2 = P(39), pcc3 = P(44), pcc4 = P(49), pconv_4 = P(54);
  const float* w5  = (const float*)d_in[59]; const float* b5  = (const float*)d_in[60];
  const float* w6  = (const float*)d_in[61]; const float* b6  = (const float*)d_in[62];
  const float* win = (const float*)d_in[63]; const float* bin = (const float*)d_in[64];
  const float* fw  = (const float*)d_in[65]; const float* fb  = (const float*)d_in[66];

  char* wsb = (char*)d_ws; size_t off = 0;
  auto alloc = [&](size_t bytes) -> void* {
    size_t a = (off + 255) & ~(size_t)255; off = a + bytes; return (void*)(wsb + a);
  };
  const size_t S16 = (size_t)Bn * C0 * 16 * 16;
  const size_t S32 = (size_t)Bn * C0 * 32 * 32;
  const size_t S64 = (size_t)Bn * C0 * 64 * 64;
  const size_t S128 = (size_t)Bn * C0 * 128 * 128;
  // bf16 NHWC activations
  __bf16* x1c    = (__bf16*)alloc(S128 * 2);
  __bf16* x2c    = (__bf16*)alloc(S64 * 2);
  __bf16* x3c    = (__bf16*)alloc(S32 * 2);
  __bf16* x4c    = (__bf16*)alloc(S16 * 2);
  __bf16* t_x3_1 = (__bf16*)alloc(S32 * 2);
  __bf16* t_a    = (__bf16*)alloc(S64 * 2);
  __bf16* t_b    = (__bf16*)alloc(S64 * 2);
  __bf16* t_x2_1 = (__bf16*)alloc(S64 * 2);
  __bf16* t_c    = (__bf16*)alloc(S128 * 2);
  __bf16* t_d    = (__bf16*)alloc(S128 * 2);
  __bf16* t_x1_1 = (__bf16*)alloc(S128 * 2);
  __bf16* t_x3_2 = (__bf16*)alloc(2 * S32 * 2);
  __bf16* t_x3_2b= (__bf16*)alloc(2 * S32 * 2);
  __bf16* t_x2_2 = (__bf16*)alloc(3 * S64 * 2);
  __bf16* t_x2_2b= (__bf16*)alloc(3 * S64 * 2);
  __bf16* t_x1_2 = (__bf16*)alloc(4 * S128 * 2);
  __bf16* t_x1_2b= (__bf16*)alloc(4 * S128 * 2);
  float*  ff1    = (float*)alloc((size_t)Bn * 16384 * 4);
  float*  X2b    = (float*)alloc((size_t)Bn * 65536 * 4);
  float*  fusedb = (float*)alloc((size_t)Bn * 2 * 16384 * 4);

  auto wpack = [&](const float* w, int Cout, int Cin, int ksq) -> __bf16* {
    __bf16* p = (__bf16*)alloc((size_t)Cout * Cin * ksq * 2);
    long tot = (long)Cout * Cin * ksq;
    repack_w<<<dim3((unsigned)((tot + 255) / 256)), 256, 0, stream>>>(w, p, Cout, Cin, ksq, tot);
    return p;
  };
  __bf16* wp1   = wpack(pconv1.w, 64, 64, 9);
  __bf16* wp2   = wpack(pconv2.w, 64, 64, 9);
  __bf16* wp3   = wpack(pconv3.w, 64, 64, 9);
  __bf16* wp7   = wpack(pconv7.w, 64, 64, 9);
  __bf16* wp4   = wpack(pconv4.w, 64, 64, 9);
  __bf16* wp5c  = wpack(pconv5.w, 128, 128, 9);
  __bf16* wp6c  = wpack(pconv6.w, 192, 192, 9);
  __bf16* wpc2  = wpack(pcc2.w, 128, 128, 9);
  __bf16* wpc3  = wpack(pcc3.w, 192, 192, 9);
  __bf16* wpc4  = wpack(pcc4.w, 256, 256, 9);
  __bf16* wpc_4 = wpack(pconv_4.w, 256, 256, 9);
  __bf16* wp_5  = wpack(w5, 64, 256, 1);

  // inputs -> bf16 NHWC
  auto cvt = [&](const float* s, __bf16* d, int HW, size_t tot) {
    k_tobf16_nhwc<<<dim3((unsigned)((tot + 255) / 256)), 256, 0, stream>>>(s, d, C0, HW, (long)tot);
  };
  cvt(x1, x1c, 128 * 128, S128);
  cvt(x2, x2c, 64 * 64, S64);
  cvt(x3, x3c, 32 * 32, S32);
  cvt(x4, x4c, 16 * 16, S16);

  auto conv = [&](const __bf16* in, const __bf16* wp, BC p, const __bf16* res,
                  __bf16* outH, float* outF,
                  int Cin, int Cout, int H, int W, int up, int ksz, int pad,
                  int ctot, int coff, int hasBN, const float* bias) {
    dim3 g((unsigned)(W / 32), (unsigned)(H / 2), (unsigned)(Bn * (Cout / 32)));
    int inH = up ? H / 2 : H, inW = up ? W / 2 : W;
    conv_wmma<<<g, 256, 0, stream>>>(in, wp,
        hasBN ? p.g : nullptr, hasBN ? p.b : bias, hasBN ? p.m : nullptr,
        hasBN ? p.v : nullptr, res, outH, outF,
        Bn, Cin, Cout, H, W, inH, inW, up, ksz, pad, ctot, coff, hasBN);
  };

  // x3_1 = BN(conv1(up(x4))) + x3
  conv(x4c, wp1, pconv1, x3c, t_x3_1, nullptr, 64, 64, 32, 32, 1, 3, 1, 64, 0, 1, nullptr);
  // x2_1 = BN(conv2(up(x3_1))) * BN(conv3(up(x3))) + x2
  conv(t_x3_1, wp2, pconv2, nullptr, t_a, nullptr, 64, 64, 64, 64, 1, 3, 1, 64, 0, 1, nullptr);
  conv(x3c,    wp3, pconv3, nullptr, t_b, nullptr, 64, 64, 64, 64, 1, 3, 1, 64, 0, 1, nullptr);
  ew_mul_add_h<<<dim3((unsigned)((S64 + 255) / 256)), 256, 0, stream>>>(t_a, t_b, x2c, t_x2_1, (long)S64);
  // x1_1 = BN(conv3(up(x2_1))) * BN(conv7(up(x2))) + x1
  conv(t_x2_1, wp3, pconv3, nullptr, t_c, nullptr, 64, 64, 128, 128, 1, 3, 1, 64, 0, 1, nullptr);
  conv(x2c,    wp7, pconv7, nullptr, t_d, nullptr, 64, 64, 128, 128, 1, 3, 1, 64, 0, 1, nullptr);
  ew_mul_add_h<<<dim3((unsigned)((S128 + 255) / 256)), 256, 0, stream>>>(t_c, t_d, x1c, t_x1_1, (long)S128);
  // x3_2 = cc2( concat[x3_1, BN(conv4(up(x4)))] )
  copy_ch_h<<<dim3((unsigned)((S32 + 255) / 256)), 256, 0, stream>>>(t_x3_1, t_x3_2, 64, 128, 0, 32 * 32, (long)S32);
  conv(x4c,    wp4,  pconv4, nullptr, t_x3_2,  nullptr, 64, 64, 32, 32, 1, 3, 1, 128, 64, 1, nullptr);
  conv(t_x3_2, wpc2, pcc2,   nullptr, t_x3_2b, nullptr, 128, 128, 32, 32, 0, 3, 1, 128, 0, 1, nullptr);
  // x2_2 = cc3( concat[x2_1, BN(conv5(up(x3_2)))] )
  copy_ch_h<<<dim3((unsigned)((S64 + 255) / 256)), 256, 0, stream>>>(t_x2_1, t_x2_2, 64, 192, 0, 64 * 64, (long)S64);
  conv(t_x3_2b, wp5c, pconv5, nullptr, t_x2_2,  nullptr, 128, 128, 64, 64, 1, 3, 1, 192, 64, 1, nullptr);
  conv(t_x2_2,  wpc3, pcc3,   nullptr, t_x2_2b, nullptr, 192, 192, 64, 64, 0, 3, 1, 192, 0, 1, nullptr);
  // x1_2 = cc4( concat[x1_1, BN(conv6(up(x2_2)))] )
  copy_ch_h<<<dim3((unsigned)((S128 + 255) / 256)), 256, 0, stream>>>(t_x1_1, t_x1_2, 64, 256, 0, 128 * 128, (long)S128);
  conv(t_x2_2b, wp6c, pconv6, nullptr, t_x1_2,  nullptr, 192, 192, 128, 128, 1, 3, 1, 256, 64, 1, nullptr);
  conv(t_x1_2,  wpc4, pcc4,   nullptr, t_x1_2b, nullptr, 256, 256, 128, 128, 0, 3, 1, 256, 0, 1, nullptr);
  // out = BN(conv_4(x1_2)); reuse t_x1_2
  __bf16* t_out = t_x1_2;
  conv(t_x1_2b, wpc_4, pconv_4, nullptr, t_out, nullptr, 256, 256, 128, 128, 0, 3, 1, 256, 0, 1, nullptr);
  // fused_features = 1x1 conv 256->64 + bias  (fp32 NCHW directly to d_out)
  float* out_ff = (float*)d_out;
  conv(t_out, wp_5, pconv_4, nullptr, nullptr, out_ff, 256, 64, 128, 128, 0, 1, 0, 64, 0, 0, b5);
  // fused_feature = 1x1 conv 64->1 + bias
  long nff = (long)Bn * 16384;
  k_conv6<<<dim3((unsigned)((nff + 255) / 256)), 256, 0, stream>>>(out_ff, w6, b6, ff1, 16384, 64, nff);
  // X = conv_in(up2x(fused_feature))
  long nX = (long)Bn * 65536;
  k_upconvin<<<dim3((unsigned)((nX + 255) / 256)), 256, 0, stream>>>(ff1, win, bin, X2b, nX);
  // haar + edge + concat
  k_haar<<<dim3((unsigned)((nff + 255) / 256)), 256, 0, stream>>>(X2b, ff1, fusedb, nff);
  // out2 (second output)
  float* out2 = (float*)d_out + S128;
  k_fusion<<<dim3((unsigned)((nff + 255) / 256)), 256, 0, stream>>>(fusedb, fw, fb, out2, nff);
}
